// ExtractLayer_54107997995555
// MI455X (gfx1250) — matrix operand order repaired and checked by reading, not compiled, not run on hardware
//
#include <hip/hip_runtime.h>
#include <hip/hip_fp16.h>

typedef __attribute__((ext_vector_type(16))) _Float16 v16h;
typedef __attribute__((ext_vector_type(8)))  _Float16 v8h;
typedef __attribute__((ext_vector_type(8)))  float    v8f;

#define DFE 128   // feature dim
#define NEG_SLOPE 0.2f
#define LN_EPS 1e-5f
#define SM_EPS 1e-16f

// ---------------------------------------------------------------- utilities

__global__ void zero_f32_kernel(float* __restrict__ p, int n) {
  int i = blockIdx.x * blockDim.x + threadIdx.x;
  if (i < n) p[i] = 0.0f;
}

__global__ void fill_u32_kernel(unsigned* __restrict__ p, int n, unsigned v) {
  int i = blockIdx.x * blockDim.x + threadIdx.x;
  if (i < n) p[i] = v;
}

__global__ void copy_f32_kernel(float* __restrict__ dst, const float* __restrict__ src, int n) {
  int i = blockIdx.x * blockDim.x + threadIdx.x;
  if (i < n) dst[i] = src[i];
}

// f32 -> f16 elementwise (vectorized x4): n multiple of 4
__global__ void cvt_f16_kernel(const float* __restrict__ src, _Float16* __restrict__ dst, int n4) {
  int i = blockIdx.x * blockDim.x + threadIdx.x;
  if (i >= n4) return;
  float4 v = *(const float4*)(src + (size_t)i * 4);
  _Float16* d = dst + (size_t)i * 4;
  d[0] = (_Float16)v.x; d[1] = (_Float16)v.y; d[2] = (_Float16)v.z; d[3] = (_Float16)v.w;
}

// Repack W[128,128] f32 row-major (k-major) into fragment-ready f16:
// Wp[((t*4 + ks)*32 + lane)*16 + i] = W[k, n],  n = t*16 + (lane&15),
//   k = ks*32 + (lane<16 ? i : 16+i)   -> each lane's 16 halves are 32B contiguous.
__global__ void pack_w_kernel(const float* __restrict__ W, _Float16* __restrict__ Wp) {
  int idx = blockIdx.x * blockDim.x + threadIdx.x;   // 0..16383
  if (idx >= 8 * 4 * 32 * 16) return;
  int i  = idx & 15;
  int l  = (idx >> 4) & 31;
  int ks = (idx >> 9) & 3;
  int t  = idx >> 11;
  int n  = t * 16 + (l & 15);
  int k  = ks * 32 + ((l < 16) ? i : 16 + i);
  Wp[idx] = (_Float16)W[(size_t)k * DFE + n];
}

// monotonic float<->uint key for atomicMax on floats
__device__ __forceinline__ unsigned f32key(float f) {
  unsigned u = __float_as_uint(f);
  return (u & 0x80000000u) ? ~u : (u | 0x80000000u);
}
__device__ __forceinline__ float keyf32(unsigned k) {
  unsigned u = (k & 0x80000000u) ? (k ^ 0x80000000u) : ~k;
  return __uint_as_float(u);
}

// ------------------------------------------------ WMMA GEMM:  Out[M,128] = Ah[M,128] @ W[128,128]
//
// Ah is pre-converted f16; Wp is the fragment-packed f16 weight (pack_w_kernel).
// One wave computes a 16x16 tile via 4 x v_wmma_f32_16x16x32_f16; per K-step the
// loads are exactly 2 b128 (A halves) + 2 b128 (B halves). Block = 8 waves = 16x128 stripe.
__global__ __launch_bounds__(256) void gemm128_wmma_kernel(
    const _Float16* __restrict__ Ah, const _Float16* __restrict__ Wp,
    float* __restrict__ Out, int M)
{
  const int lane = threadIdx.x & 31;
  const int wave = threadIdx.x >> 5;      // = column tile t
  const int row0 = blockIdx.x * 16;
  const int n0   = wave * 16;
  const bool lo  = lane < 16;
  const int l15  = lane & 15;

  int mrow = row0 + l15;
  int mcl  = (mrow < M) ? mrow : (M - 1);          // clamp loads, mask stores
  const _Float16* __restrict__ arow  = Ah + (size_t)mcl * DFE;
  const _Float16* __restrict__ bbase = Wp + ((size_t)(wave * 4) * 32 + lane) * 16;

  v8f c = {};
#pragma unroll
  for (int ks = 0; ks < 4; ++ks) {
    const int k0 = ks * 32;
    // A fragment: lanes 0-15 hold K=[k0..k0+15]; lanes 16-31 hold K=[k0+16..k0+31]
    const int ka = k0 + (lo ? 0 : 16);
    const int kb = k0 + (lo ? 8 : 24);
    v8h alo = *(const v8h*)(arow + ka);
    v8h ahi = *(const v8h*)(arow + kb);
    v16h a = __builtin_shufflevector(alo, ahi,
                                     0, 1, 2, 3, 4, 5, 6, 7, 8, 9, 10, 11, 12, 13, 14, 15);
    // B fragment: 16 contiguous halves per lane from the packed weight
    const v8h* bp = (const v8h*)(bbase + (size_t)ks * (32 * 16));
    v16h b = __builtin_shufflevector(bp[0], bp[1],
                                     0, 1, 2, 3, 4, 5, 6, 7, 8, 9, 10, 11, 12, 13, 14, 15);

    c = __builtin_amdgcn_wmma_f32_16x16x32_f16(
        /*neg_a=*/false, a, /*neg_b=*/false, b,
        /*c_mod=*/(short)0, c, /*reuse_a=*/false, /*reuse_b=*/false);
  }

  // store D: VGPR j -> M = row0 + j (+8 for upper half-wave), N = n0 + (lane&15)
  const int nn = n0 + l15;
  const int mbase = row0 + (lo ? 0 : 8);
#pragma unroll
  for (int j = 0; j < 8; ++j) {
    int mm = mbase + j;
    if (mm < M) Out[(size_t)mm * DFE + nn] = c[j];
  }
}

// ------------------------------------------------ edge logits + per-dst max (wave per edge)
__global__ __launch_bounds__(256) void edge_logits_kernel(
    const float* __restrict__ HL, const float* __restrict__ HR,
    const int* __restrict__ src, const int* __restrict__ dst, int E,
    const float* __restrict__ att,
    const float* __restrict__ ea, const float* __restrict__ We,   // may be null
    float* __restrict__ logit, unsigned* __restrict__ maxb)
{
  const int e = (blockIdx.x * blockDim.x + threadIdx.x) >> 5;   // uniform per wave
  const int lane = threadIdx.x & 31;
  if (e >= E) return;
  const int s = src[e], d = dst[e];

  float4 m  = *(const float4*)(HL + (size_t)s * DFE + lane * 4);
  float4 hr = *(const float4*)(HR + (size_t)d * DFE + lane * 4);
  m.x += hr.x; m.y += hr.y; m.z += hr.z; m.w += hr.w;

  if (ea != nullptr) {        // fused edge-feature transform: msg += ea[e] @ We (8 x 128)
#pragma unroll
    for (int k = 0; k < 8; ++k) {
      float ev = ea[(size_t)e * 8 + k];
      float4 w = *(const float4*)(We + (size_t)k * DFE + lane * 4);
      m.x += ev * w.x; m.y += ev * w.y; m.z += ev * w.z; m.w += ev * w.w;
    }
  }

  const float4 a4 = *(const float4*)(att + lane * 4);
  float lx = (m.x > 0.f) ? m.x : NEG_SLOPE * m.x;
  float ly = (m.y > 0.f) ? m.y : NEG_SLOPE * m.y;
  float lz = (m.z > 0.f) ? m.z : NEG_SLOPE * m.z;
  float lw = (m.w > 0.f) ? m.w : NEG_SLOPE * m.w;
  float v = lx * a4.x + ly * a4.y + lz * a4.z + lw * a4.w;
#pragma unroll
  for (int off = 16; off > 0; off >>= 1) v += __shfl_xor(v, off, 32);

  if (lane == 0) {
    logit[e] = v;
    atomicMax(&maxb[d], f32key(v));
  }
}

// ------------------------------------------------ exp + per-dst denom (thread per edge)
__global__ __launch_bounds__(256) void edge_exp_kernel(
    const float* __restrict__ logit, const int* __restrict__ dst, int E,
    const unsigned* __restrict__ maxb, float* __restrict__ eexp,
    float* __restrict__ denom)
{
  int e = blockIdx.x * blockDim.x + threadIdx.x;
  if (e >= E) return;
  int d = dst[e];
  float ex = __expf(logit[e] - keyf32(maxb[d]));
  eexp[e] = ex;
  atomicAdd(&denom[d], ex);
}

// ------------------------------------------------ alpha-weighted scatter (wave per edge)
__global__ __launch_bounds__(256) void edge_scatter_kernel(
    const float* __restrict__ HL,
    const int* __restrict__ src, const int* __restrict__ dst, int E,
    const float* __restrict__ eexp, const float* __restrict__ denom,
    float* __restrict__ acc)
{
  const int e = (blockIdx.x * blockDim.x + threadIdx.x) >> 5;
  const int lane = threadIdx.x & 31;
  if (e >= E) return;
  const int s = src[e], d = dst[e];
  const float alpha = eexp[e] / (denom[d] + SM_EPS);
  float4 hl = *(const float4*)(HL + (size_t)s * DFE + lane * 4);
  float* ap = acc + (size_t)d * DFE + lane * 4;
  atomicAdd(ap + 0, alpha * hl.x);
  atomicAdd(ap + 1, alpha * hl.y);
  atomicAdd(ap + 2, alpha * hl.z);
  atomicAdd(ap + 3, alpha * hl.w);
}

// ------------------------------------------------ bias + LayerNorm + tanh, add into out (wave per node)
__global__ __launch_bounds__(256) void node_finalize_kernel(
    const float* __restrict__ acc, const float* __restrict__ bias,
    const float* __restrict__ gamma, const float* __restrict__ beta,
    float* __restrict__ out, int N)
{
  const int n = (blockIdx.x * blockDim.x + threadIdx.x) >> 5;
  const int lane = threadIdx.x & 31;
  if (n >= N) return;

  float4 v  = *(const float4*)(acc + (size_t)n * DFE + lane * 4);
  float4 b4 = *(const float4*)(bias + lane * 4);
  v.x += b4.x; v.y += b4.y; v.z += b4.z; v.w += b4.w;

  float s = v.x + v.y + v.z + v.w;
#pragma unroll
  for (int off = 16; off > 0; off >>= 1) s += __shfl_xor(s, off, 32);
  const float mu = s * (1.0f / DFE);

  float dx = v.x - mu, dy = v.y - mu, dz = v.z - mu, dw = v.w - mu;
  float q = dx * dx + dy * dy + dz * dz + dw * dw;
#pragma unroll
  for (int off = 16; off > 0; off >>= 1) q += __shfl_xor(q, off, 32);
  const float inv = rsqrtf(q * (1.0f / DFE) + LN_EPS);

  float4 g4 = *(const float4*)(gamma + lane * 4);
  float4 e4 = *(const float4*)(beta + lane * 4);
  float* op = out + (size_t)n * DFE + lane * 4;
  float4 o = *(float4*)op;
  o.x += tanhf(dx * inv * g4.x + e4.x);
  o.y += tanhf(dy * inv * g4.y + e4.y);
  o.z += tanhf(dz * inv * g4.z + e4.z);
  o.w += tanhf(dw * inv * g4.w + e4.w);
  *(float4*)op = o;
}

// ---------------------------------------------------------------- host driver

static inline int cdiv(int a, int b) { return (a + b - 1) / b; }

extern "C" void kernel_launch(void* const* d_in, const int* in_sizes, int n_in,
                              void* d_out, int out_size, void* d_ws, size_t ws_size,
                              hipStream_t stream) {
  (void)in_sizes; (void)n_in; (void)out_size; (void)ws_size;

  const int N_OP = 50000, N_MACH = 5000, N_AGV = 5000;

  // Inputs flattened in setup_inputs() insertion order (nested params dicts too):
  // 0 x_op, 1 x_mach, 2 x_agv, 3 proc_ea,
  // 4..9   pred : Wl Wr att b ln_g ln_b
  // 10..15 succ, 16..22 proc (+We last), 23..28 wait, 29..34 am, 35..40 ma,
  // 41 pred_ei [2,200000], 42 succ_ei, 43 proc_ei [2,400000], 44 wait_ei, 45 am_ei [2,20000], 46 ma_ei
  const float* x_op    = (const float*)d_in[0];
  const float* x_mach  = (const float*)d_in[1];
  const float* x_agv   = (const float*)d_in[2];
  const float* proc_ea = (const float*)d_in[3];

  float* out = (float*)d_out;
  float* ws  = (float*)d_ws;

  // workspace layout (float units), reused across relations (processed sequentially)
  float*    HL   = ws;                          // 50000*128
  float*    HR   = ws + 6400000;                // 50000*128
  float*    ACC  = ws + 12800000;               // 50000*128
  float*    LOG  = ws + 19200000;               // 400000
  float*    EXPB = ws + 19600000;               // 400000
  float*    DEN  = ws + 20000000;               // 50000
  unsigned* MAXB = (unsigned*)(ws + 20050000);  // 50000
  _Float16* XH_OP   = (_Float16*)(ws + 20100000);  // 50000*128 f16 (3.2M floats)
  _Float16* XH_MACH = (_Float16*)(ws + 23300000);  // 5000*128 f16
  _Float16* XH_AGV  = (_Float16*)(ws + 23620000);  // 5000*128 f16
  _Float16* WLP     = (_Float16*)(ws + 23940000);  // 16384 halves
  _Float16* WRP     = (_Float16*)(ws + 23948192);  // 16384 halves

  // one-time f16 conversion of node features (reused by all relations)
  cvt_f16_kernel<<<cdiv(N_OP * DFE / 4, 256), 256, 0, stream>>>(x_op, XH_OP, N_OP * DFE / 4);
  cvt_f16_kernel<<<cdiv(N_MACH * DFE / 4, 256), 256, 0, stream>>>(x_mach, XH_MACH, N_MACH * DFE / 4);
  cvt_f16_kernel<<<cdiv(N_AGV * DFE / 4, 256), 256, 0, stream>>>(x_agv, XH_AGV, N_AGV * DFE / 4);

  // residual init: out = concat(x_op, x_mach, x_agv)
  copy_f32_kernel<<<cdiv(N_OP * DFE, 256), 256, 0, stream>>>(out, x_op, N_OP * DFE);
  copy_f32_kernel<<<cdiv(N_MACH * DFE, 256), 256, 0, stream>>>(out + 6400000, x_mach, N_MACH * DFE);
  copy_f32_kernel<<<cdiv(N_AGV * DFE, 256), 256, 0, stream>>>(out + 7040000, x_agv, N_AGV * DFE);

  struct RelCfg {
    int pb;            // first param index (Wl)
    int eiIdx;         // edge index input
    int E;
    int Ns, Nd;
    const _Float16 *XHs, *XHd;
    size_t outOff;
    bool hasEdgeFeat;
  };
  const RelCfg rels[6] = {
    { 4, 41, 200000, N_OP,   N_OP,   XH_OP,   XH_OP,   0,       false},  // pred
    {10, 42, 200000, N_OP,   N_OP,   XH_OP,   XH_OP,   0,       false},  // succ
    {16, 43, 400000, N_MACH, N_OP,   XH_MACH, XH_OP,   0,       true },  // proc
    {23, 44, 400000, N_OP,   N_MACH, XH_OP,   XH_MACH, 6400000, false},  // wait
    {29, 45, 20000,  N_AGV,  N_MACH, XH_AGV,  XH_MACH, 6400000, false},  // am
    {35, 46, 20000,  N_MACH, N_AGV,  XH_MACH, XH_AGV,  7040000, false},  // ma
  };

  for (int r = 0; r < 6; ++r) {
    const RelCfg& R = rels[r];
    const float* Wl   = (const float*)d_in[R.pb + 0];
    const float* Wr   = (const float*)d_in[R.pb + 1];
    const float* att  = (const float*)d_in[R.pb + 2];
    const float* bias = (const float*)d_in[R.pb + 3];
    const float* ln_g = (const float*)d_in[R.pb + 4];
    const float* ln_b = (const float*)d_in[R.pb + 5];
    const float* We   = R.hasEdgeFeat ? (const float*)d_in[R.pb + 6] : nullptr;
    const float* ea   = R.hasEdgeFeat ? proc_ea : nullptr;
    const int*   ei   = (const int*)d_in[R.eiIdx];
    const int*   src  = ei;
    const int*   dst  = ei + R.E;

    // pack weights into fragment-ready f16 layout
    pack_w_kernel<<<64, 256, 0, stream>>>(Wl, WLP);
    pack_w_kernel<<<64, 256, 0, stream>>>(Wr, WRP);

    // linear transforms (WMMA)
    gemm128_wmma_kernel<<<cdiv(R.Ns, 16), 256, 0, stream>>>(R.XHs, WLP, HL, R.Ns);
    gemm128_wmma_kernel<<<cdiv(R.Nd, 16), 256, 0, stream>>>(R.XHd, WRP, HR, R.Nd);

    // reset per-relation accumulators
    zero_f32_kernel<<<cdiv(R.Nd * DFE, 256), 256, 0, stream>>>(ACC, R.Nd * DFE);
    zero_f32_kernel<<<cdiv(R.Nd, 256), 256, 0, stream>>>(DEN, R.Nd);
    fill_u32_kernel<<<cdiv(R.Nd, 256), 256, 0, stream>>>(MAXB, R.Nd, 0u);

    // attention + softmax + aggregate
    edge_logits_kernel<<<cdiv(R.E * 32, 256), 256, 0, stream>>>(
        HL, HR, src, dst, R.E, att, ea, We, LOG, MAXB);
    edge_exp_kernel<<<cdiv(R.E, 256), 256, 0, stream>>>(
        LOG, dst, R.E, MAXB, EXPB, DEN);
    edge_scatter_kernel<<<cdiv(R.E * 32, 256), 256, 0, stream>>>(
        HL, src, dst, R.E, EXPB, DEN, ACC);

    // bias + LayerNorm + tanh, accumulate into output slice
    node_finalize_kernel<<<cdiv(R.Nd * 32, 256), 256, 0, stream>>>(
        ACC, bias, ln_g, ln_b, out + R.outOff, R.Nd);
  }
}
